// TransformerDecoderLayer_53429393162551
// MI455X (gfx1250) — matrix operand order repaired and checked by reading, not compiled, not run on hardware
//
#include <hip/hip_runtime.h>

typedef _Float16 h16;
typedef __attribute__((ext_vector_type(8)))  _Float16 v8h;
typedef __attribute__((ext_vector_type(16))) _Float16 v16h;
typedef __attribute__((ext_vector_type(8)))  float    v8f;

#define ASTR 40   // LDS A-tile row stride in halves (80B -> conflict-free b128 reads)
#define KSTR 258  // LDS K/V row stride in halves for CA kernel

__device__ __forceinline__ v8f wmma_f16(v16h a, v16h b, v8f c) {
  return __builtin_amdgcn_wmma_f32_16x16x32_f16(false, a, false, b, (short)0, c,
                                                false, false);
}

__device__ __forceinline__ v8f zero8() {
  v8f z;
#pragma unroll
  for (int i = 0; i < 8; ++i) z[i] = 0.f;
  return z;
}

__device__ __forceinline__ v16h cat8(v8h lo, v8h hi) {
  v16h r;
#pragma unroll
  for (int i = 0; i < 8; ++i) { r[i] = lo[i]; r[i + 8] = hi[i]; }
  return r;
}

// A operand (16x32 f16), row-major buffer with stride (halves).
// lane<16: rows 0-15, K = {0..7, 16..23}; lane>=16: same rows, K = {8..15, 24..31}
__device__ __forceinline__ v16h ld_a_lds(const h16* base, int stride, int row0,
                                         int k0, int lane) {
  const h16* p = base + (size_t)(row0 + (lane & 15)) * stride + k0 + ((lane & 16) ? 8 : 0);
  return cat8(*(const v8h*)p, *(const v8h*)(p + 16));
}

// B operand (32x16 f16) from a row-major (N,K) buffer: lane holds column n0+lane%16,
// lower lane-half K = {0..7, 16..23}, upper half K = {8..15, 24..31}.
__device__ __forceinline__ v16h ld_b(const h16* W, int ldk, int n0, int k0, int lane) {
  const h16* p = W + (size_t)(n0 + (lane & 15)) * ldk + k0 + ((lane & 16) ? 8 : 0);
  return cat8(*(const v8h*)p, *(const v8h*)(p + 16));
}

// Async global->LDS copy of 16B (gfx1250 TDM-adjacent path, ASYNCcnt-tracked).
// Generic LDS pointers carry the group-segment byte offset in their low 32 bits
//; inst offset is added to BOTH addresses.
__device__ __forceinline__ void async_cp16(void* lds_dst, const void* gsrc) {
  unsigned loff = (unsigned)(uintptr_t)lds_dst;
  unsigned long long g = (unsigned long long)(uintptr_t)gsrc;
  asm volatile("global_load_async_to_lds_b128 %0, %1, off"
               :: "v"(loff), "v"(g) : "memory");
}
__device__ __forceinline__ void async_cp16_off16(void* lds_dst, const void* gsrc) {
  unsigned loff = (unsigned)(uintptr_t)lds_dst;
  unsigned long long g = (unsigned long long)(uintptr_t)gsrc;
  asm volatile("global_load_async_to_lds_b128 %0, %1, off offset:16"
               :: "v"(loff), "v"(g) : "memory");
}
__device__ __forceinline__ void async_wait0() {
  asm volatile("s_wait_asynccnt 0x0" ::: "memory");
}

// ---------------- weight folding: dst[n][k] = sum_j Wo[n][j] * Wi[j][k] (f16 out)
__global__ __launch_bounds__(256) void k_fold_w(const float* __restrict__ Wo,
                                                const float* __restrict__ Wi,
                                                h16* __restrict__ dst) {
  const int n = blockIdx.x, k = threadIdx.x;
  float s = 0.f;
  for (int j = 0; j < 256; ++j) s += Wo[n * 256 + j] * Wi[j * 256 + k];
  dst[n * 256 + k] = (h16)s;
}

__global__ __launch_bounds__(256) void k_cvt_h(const float* __restrict__ s,
                                               h16* __restrict__ d, int n) {
  int i = blockIdx.x * 256 + threadIdx.x;
  if (i < n) d[i] = (h16)s[i];
}

// bias fold: dst[n] = (init ? bo[n] : dst[n]) + sum_j Wo[n][j]*bi[j]
__global__ __launch_bounds__(256) void k_fold_b(const float* __restrict__ Wo,
                                                const float* __restrict__ bi,
                                                const float* __restrict__ bo,
                                                float* __restrict__ dst, int init) {
  const int n = threadIdx.x;
  float s = init ? (bo ? bo[n] : 0.f) : dst[n];
  for (int j = 0; j < 256; ++j) s += Wo[n * 256 + j] * bi[j];
  dst[n] = s;
}

// ---------------- generic WMMA GEMM: C[M,N] = sum_t A_t[M,K] @ B_t[N,K]^T + bias
// 256 threads = 8 waves in 4x2; block tile 128x64; each wave 32x32 (2x2 wmma tiles)
// a0h==1: A is f16 in global; staged via global_load_async_to_lds_b128 (no VGPR trip)
__global__ __launch_bounds__(256, 1) void k_gemm(
    const void* A0v, const float* A1, const float* A2,
    const h16* B0, const h16* B1, const h16* B2,
    int nterms, int M, int N, int K,
    const float* __restrict__ bias, int relu,
    float* outF, h16* outH, int a0h) {
  __shared__ h16 As[3][128 * ASTR];
  const int tid = threadIdx.x, lane = tid & 31, wave = tid >> 5;
  const int wm = wave >> 1, wn = wave & 1;
  const int bm = blockIdx.x * 128, bn = blockIdx.y * 64;

  v8f acc[2][2];
#pragma unroll
  for (int mi = 0; mi < 2; ++mi)
#pragma unroll
    for (int ni = 0; ni < 2; ++ni) acc[mi][ni] = zero8();

  const int r = tid >> 1;
  const int cs = (tid & 1) << 4;

  for (int k0 = 0; k0 < K; k0 += 32) {
    for (int t = 0; t < nterms; ++t) {
      h16* dst = &As[t][r * ASTR + cs];
      const int grow = bm + r;
      if (t == 0 && a0h) {
        // M is a multiple of 128 on all a0h call sites -> no row guard needed
        const h16* src = (const h16*)A0v + (size_t)grow * K + k0 + cs;
        async_cp16(dst, src);
        async_cp16_off16(dst, src);
      } else if (grow < M) {
        const float* srcf = (t == 0 ? (const float*)A0v : (t == 1 ? A1 : A2));
        const float* src = srcf + (size_t)grow * K + k0 + cs;
#pragma unroll
        for (int i = 0; i < 16; ++i) dst[i] = (h16)src[i];
      } else {
#pragma unroll
        for (int i = 0; i < 16; ++i) dst[i] = (h16)0.f;
      }
    }
    if (a0h) async_wait0();
    __syncthreads();
    for (int t = 0; t < nterms; ++t) {
      const h16* Bt = (t == 0 ? B0 : (t == 1 ? B1 : B2));
#pragma unroll
      for (int mi = 0; mi < 2; ++mi) {
        v16h a = ld_a_lds(&As[t][0], ASTR, wm * 32 + mi * 16, 0, lane);
#pragma unroll
        for (int ni = 0; ni < 2; ++ni) {
          v16h b = ld_b(Bt, K, bn + wn * 32 + ni * 16, k0, lane);
          acc[mi][ni] = wmma_f16(a, b, acc[mi][ni]);
        }
      }
    }
    __syncthreads();
  }
#pragma unroll
  for (int mi = 0; mi < 2; ++mi)
#pragma unroll
    for (int ni = 0; ni < 2; ++ni)
#pragma unroll
      for (int v = 0; v < 8; ++v) {
        int row = bm + wm * 32 + mi * 16 + v + ((lane & 16) ? 8 : 0);
        int col = bn + wn * 32 + ni * 16 + (lane & 15);
        if (row < M) {
          float val = acc[mi][ni][v] + bias[col];
          if (relu) val = fmaxf(val, 0.f);
          if (outF) outF[(size_t)row * N + col] = val;
          if (outH) outH[(size_t)row * N + col] = (h16)val;
        }
      }
}

// ---------------- self-attention: one wave per (batch n, head h).
// Q,K,V f16 (1024,256); scores 64x64 via WMMA, softmax, P@V via WMMA.
// Q/K tiles staged with async global->LDS copies (rows are 64B contiguous).
__global__ __launch_bounds__(32, 1) void k_sa_attn(
    const h16* __restrict__ Q, const h16* __restrict__ Kb, const h16* __restrict__ Vb,
    float* __restrict__ attn_ws, h16* __restrict__ attn_out) {
  __shared__ h16 Qs[64 * ASTR];
  __shared__ h16 Ks[64 * ASTR];
  __shared__ h16 Vt[32 * 72];   // Vt[d][s]
  __shared__ float S[64 * 68];
  __shared__ h16 P[64 * 72];
  __shared__ float rowsum[64];
  const int n = blockIdx.x, h = blockIdx.y, lane = threadIdx.x;

  for (int t = lane; t < 64; t += 32) {
    const size_t m = (size_t)(t * 16 + n) * 256 + h * 32;
#pragma unroll
    for (int c = 0; c < 32; c += 8) {   // 4 x 16B async copies per row
      async_cp16(&Qs[t * ASTR + c], Q + m + c);
      async_cp16(&Ks[t * ASTR + c], Kb + m + c);
    }
#pragma unroll
    for (int c = 0; c < 32; ++c) Vt[c * 72 + t] = Vb[m + c];
  }
  async_wait0();
  __syncthreads();

  const float scale = 0.1767766952966369f;  // 1/sqrt(32)
#pragma unroll
  for (int mi = 0; mi < 4; ++mi) {
    v16h a = ld_a_lds(Qs, ASTR, mi * 16, 0, lane);
#pragma unroll
    for (int ni = 0; ni < 4; ++ni) {
      v16h b = ld_b(Ks, ASTR, ni * 16, 0, lane);
      v8f acc = zero8();
      acc = wmma_f16(a, b, acc);
#pragma unroll
      for (int v = 0; v < 8; ++v) {
        int rr = mi * 16 + v + ((lane & 16) ? 8 : 0);
        int ss = ni * 16 + (lane & 15);
        S[rr * 68 + ss] = acc[v] * scale;
      }
    }
  }
  __syncthreads();

  for (int rsel = 0; rsel < 2; ++rsel) {
    const int rr = lane + rsel * 32;
    float mx = -1e30f;
    for (int s2 = 0; s2 < 64; ++s2) mx = fmaxf(mx, S[rr * 68 + s2]);
    float sum = 0.f;
    for (int s2 = 0; s2 < 64; ++s2) {
      float e = __expf(S[rr * 68 + s2] - mx);
      P[rr * 72 + s2] = (h16)e;
      sum += e;
    }
    rowsum[rr] = sum;
    const float inv = 1.f / sum;
    float* aw = attn_ws + (((size_t)n * 8 + h) * 64 + rr) * 64;
    for (int s2 = 0; s2 < 64; ++s2)
      aw[s2] = __expf(S[rr * 68 + s2] - mx) * inv;
  }
  __syncthreads();

#pragma unroll
  for (int mi = 0; mi < 4; ++mi)
#pragma unroll
    for (int di = 0; di < 2; ++di) {
      v8f acc = zero8();
#pragma unroll
      for (int ks = 0; ks < 2; ++ks) {
        v16h a = ld_a_lds(P, 72, mi * 16, ks * 32, lane);
        v16h b = ld_b(Vt, 72, di * 16, ks * 32, lane);
        acc = wmma_f16(a, b, acc);
      }
#pragma unroll
      for (int v = 0; v < 8; ++v) {
        int rr = mi * 16 + v + ((lane & 16) ? 8 : 0);
        int dd = di * 16 + (lane & 15);
        attn_out[(size_t)(rr * 16 + n) * 256 + h * 32 + dd] = (h16)(acc[v] / rowsum[rr]);
      }
    }
}

// weights = attn.mean(heads): (16,8,64,64) -> (16,64,64)
__global__ __launch_bounds__(256) void k_wred(const float* __restrict__ aw,
                                              float* __restrict__ wo) {
  int idx = blockIdx.x * 256 + threadIdx.x;
  int n = idx >> 12, ls = idx & 4095;
  float s = 0.f;
#pragma unroll
  for (int hh = 0; hh < 8; ++hh) s += aw[(((size_t)n * 8 + hh) << 12) + ls];
  wo[idx] = 0.125f * s;
}

// ---------------- fused cross-attention: one block per query j (512 blocks, 8 waves).
// Streams 64-key chunks: WMMA-projects mem/pos -> K,V in LDS, then per-head (wave h)
// online-softmax accumulation. Avoids materializing the (100352,256) K/V tensors.
__global__ __launch_bounds__(256, 1) void k_ca_attn(
    const float* __restrict__ mem, const float* __restrict__ pos,
    const float* __restrict__ caq,
    const h16* __restrict__ Wk0, const h16* __restrict__ Wk1, const h16* __restrict__ Wv,
    const float* __restrict__ bk, const float* __restrict__ bv,
    const int* __restrict__ dur, h16* __restrict__ out) {
  __shared__ float q_sh[256];
  __shared__ h16 Am[64 * ASTR];
  __shared__ h16 Ap[64 * ASTR];
  __shared__ h16 Ksh[64 * KSTR];
  __shared__ h16 Vsh[64 * KSTR];
  __shared__ float p_sh[8 * 64];
  const int j = blockIdx.x;
  const int tid = threadIdx.x, lane = tid & 31, wave = tid >> 5;

  int st = 0, bsel = 15;
  for (int i = 0; i < 16; ++i) {
    int d = dur[i];
    if (j < st + d) { bsel = i; break; }
    st += d;
  }
  const int qrow = (j - st) * 16 + bsel;
  q_sh[tid] = caq[(size_t)qrow * 256 + tid];
  __syncthreads();

  float m_run = -1e30f, s_run = 0.f, o_run = 0.f;
  const int h = wave;
  const float scale = 0.1767766952966369f;

  const int r = tid >> 2;
  const int cs = (tid & 3) << 3;

  for (int chunk = 0; chunk < 4; ++chunk) {
    const int s_base = chunk * 64;
    v8f accK[4][2], accV[4][2];
#pragma unroll
    for (int mi = 0; mi < 4; ++mi)
#pragma unroll
      for (int ni = 0; ni < 2; ++ni) { accK[mi][ni] = zero8(); accV[mi][ni] = zero8(); }

    const int srow = s_base + r;
    const int sclamp = srow < 196 ? srow : 195;
    const size_t gbase = ((size_t)sclamp * 512 + j) * 256 + cs;

    for (int kst = 0; kst < 8; ++kst) {
      const int k0 = kst * 32;
      {
        const float* ms = mem + gbase + k0;
        const float* ps = pos + gbase + k0;
        h16* dm = &Am[r * ASTR + cs];
        h16* dp = &Ap[r * ASTR + cs];
#pragma unroll
        for (int i = 0; i < 8; ++i) { dm[i] = (h16)ms[i]; dp[i] = (h16)ps[i]; }
      }
      __syncthreads();
#pragma unroll
      for (int mi = 0; mi < 4; ++mi) {
        v16h am = ld_a_lds(Am, ASTR, mi * 16, 0, lane);
        v16h ap = ld_a_lds(Ap, ASTR, mi * 16, 0, lane);
#pragma unroll
        for (int ni = 0; ni < 2; ++ni) {
          const int nc = wave * 32 + ni * 16;
          accK[mi][ni] = wmma_f16(am, ld_b(Wk0, 256, nc, k0, lane), accK[mi][ni]);
          accK[mi][ni] = wmma_f16(ap, ld_b(Wk1, 256, nc, k0, lane), accK[mi][ni]);
          accV[mi][ni] = wmma_f16(am, ld_b(Wv, 256, nc, k0, lane), accV[mi][ni]);
        }
      }
      __syncthreads();
    }
    // dump this chunk's K,V (with folded biases) to LDS f16
#pragma unroll
    for (int mi = 0; mi < 4; ++mi)
#pragma unroll
      for (int ni = 0; ni < 2; ++ni)
#pragma unroll
        for (int v = 0; v < 8; ++v) {
          int rr = mi * 16 + v + ((lane & 16) ? 8 : 0);
          int cc = wave * 32 + ni * 16 + (lane & 15);
          Ksh[rr * KSTR + cc] = (h16)(accK[mi][ni][v] + bk[cc]);
          Vsh[rr * KSTR + cc] = (h16)(accV[mi][ni][v] + bv[cc]);
        }
    __syncthreads();

    // online softmax update; wave h owns head h; 2 keys per lane
    float d0 = 0.f, d1 = 0.f;
#pragma unroll
    for (int dd = 0; dd < 32; ++dd) {
      float qd = q_sh[h * 32 + dd];
      d0 += qd * (float)Ksh[lane * KSTR + h * 32 + dd];
      d1 += qd * (float)Ksh[(lane + 32) * KSTR + h * 32 + dd];
    }
    float sc0 = (s_base + lane < 196) ? d0 * scale : -1e30f;
    float sc1 = (s_base + lane + 32 < 196) ? d1 * scale : -1e30f;
    float mx = fmaxf(sc0, sc1);
    for (int o = 16; o >= 1; o >>= 1) mx = fmaxf(mx, __shfl_xor(mx, o, 32));
    float m_new = fmaxf(m_run, mx);
    float fac = __expf(m_run - m_new);
    float e0 = __expf(sc0 - m_new);
    float e1 = __expf(sc1 - m_new);
    p_sh[h * 64 + lane] = e0;
    p_sh[h * 64 + lane + 32] = e1;
    float ls = e0 + e1;
    for (int o = 16; o >= 1; o >>= 1) ls += __shfl_xor(ls, o, 32);
    s_run = s_run * fac + ls;
    o_run *= fac;
    for (int kk = 0; kk < 64; ++kk)
      o_run += p_sh[h * 64 + kk] * (float)Vsh[kk * KSTR + h * 32 + lane];
    m_run = m_new;
    __syncthreads();
  }
  out[(size_t)j * 256 + h * 32 + lane] = (h16)(o_run / s_run);
}

// ---------------- residual + layernorm; mode 1 gathers CA output by (start[b]+t)
__global__ __launch_bounds__(256) void k_add_ln(
    const float* __restrict__ x, const float* __restrict__ r,
    const float* __restrict__ g, const float* __restrict__ bb,
    float* outF, h16* outH, const int* dur, int mode) {
  const int row = blockIdx.x, tid = threadIdx.x;
  float v = x[(size_t)row * 256 + tid];
  if (mode == 0) {
    v += r[(size_t)row * 256 + tid];
  } else {
    const int t = row >> 4, b = row & 15;
    int st = 0;
    for (int i = 0; i < b; ++i) st += dur[i];
    if (t < dur[b]) v += r[(size_t)(st + t) * 256 + tid];
  }
  __shared__ float red[256];
  red[tid] = v; __syncthreads();
  for (int o = 128; o > 0; o >>= 1) { if (tid < o) red[tid] += red[tid + o]; __syncthreads(); }
  const float mean = red[0] * (1.f / 256.f);
  __syncthreads();
  const float dd = v - mean;
  red[tid] = dd * dd; __syncthreads();
  for (int o = 128; o > 0; o >>= 1) { if (tid < o) red[tid] += red[tid + o]; __syncthreads(); }
  const float var = red[0] * (1.f / 256.f);
  const float y = dd * rsqrtf(var + 1e-5f) * g[tid] + bb[tid];
  if (outF) outF[(size_t)row * 256 + tid] = y;
  if (outH) outH[(size_t)row * 256 + tid] = (h16)y;
}

extern "C" void kernel_launch(void* const* d_in, const int* in_sizes, int n_in,
                              void* d_out, int out_size, void* d_ws, size_t ws_size,
                              hipStream_t stream) {
  (void)in_sizes; (void)n_in; (void)out_size; (void)ws_size;
  // top-level inputs in setup_inputs() insertion order; params flattened with
  // sorted keys at every dict level (jax pytree convention)
  const float* tgt   = (const float*)d_in[0];
  const float* mem   = (const float*)d_in[1];
  const float* pos   = (const float*)d_in[2];
  const float* qpos  = (const float*)d_in[3];
  const float* qtime = (const float*)d_in[4];
  const float* qsine = (const float*)d_in[5];
  const float *ca_bk = (const float*)d_in[6],  *ca_bo = (const float*)d_in[7];
  const float *ca_bq = (const float*)d_in[8],  *ca_bv = (const float*)d_in[9];
  const float *ca_wk = (const float*)d_in[10], *ca_wo = (const float*)d_in[11];
  const float *ca_wq = (const float*)d_in[12], *ca_wv = (const float*)d_in[13];
  const float *ca_kc_w = (const float*)d_in[14], *ca_kc_b = (const float*)d_in[15];
  const float *ca_kp_w = (const float*)d_in[16], *ca_kp_b = (const float*)d_in[17];
  const float *ca_qc_w = (const float*)d_in[18], *ca_qc_b = (const float*)d_in[19];
  const float *ca_qs_w = (const float*)d_in[20], *ca_qs_b = (const float*)d_in[21];
  const float *ca_qt_w = (const float*)d_in[22], *ca_qt_b = (const float*)d_in[23];
  const float *ca_v_w  = (const float*)d_in[24], *ca_v_b  = (const float*)d_in[25];
  const float *lin1_w = (const float*)d_in[26], *lin1_b = (const float*)d_in[27];
  const float *lin2_w = (const float*)d_in[28], *lin2_b = (const float*)d_in[29];
  const float *n1_b = (const float*)d_in[30], *n1_g = (const float*)d_in[31];
  const float *n3_b = (const float*)d_in[32], *n3_g = (const float*)d_in[33];
  const float *n4_b = (const float*)d_in[34], *n4_g = (const float*)d_in[35];
  const float *sa_bk = (const float*)d_in[36], *sa_bo = (const float*)d_in[37];
  const float *sa_bq = (const float*)d_in[38], *sa_bv = (const float*)d_in[39];
  const float *sa_wk = (const float*)d_in[40], *sa_wo = (const float*)d_in[41];
  const float *sa_wq = (const float*)d_in[42], *sa_wv = (const float*)d_in[43];
  const float *sa_kc_w = (const float*)d_in[44], *sa_kc_b = (const float*)d_in[45];
  const float *sa_kp_w = (const float*)d_in[46], *sa_kp_b = (const float*)d_in[47];
  const float *sa_kt_w = (const float*)d_in[48], *sa_kt_b = (const float*)d_in[49];
  const float *sa_qc_w = (const float*)d_in[50], *sa_qc_b = (const float*)d_in[51];
  const float *sa_qp_w = (const float*)d_in[52], *sa_qp_b = (const float*)d_in[53];
  const float *sa_qt_w = (const float*)d_in[54], *sa_qt_b = (const float*)d_in[55];
  const float *sa_v_w  = (const float*)d_in[56], *sa_v_b  = (const float*)d_in[57];
  const int* dur = (const int*)d_in[58];

  char* w = (char*)d_ws;
  size_t off = 0;
  auto A = [&](size_t n) -> void* {
    void* p = w + off;
    off += (n + 255) & ~(size_t)255;
    return p;
  };
  const size_t FW = 256 * 256 * sizeof(h16);
  h16 *Wsa_q0 = (h16*)A(FW), *Wsa_q1 = (h16*)A(FW), *Wsa_q2 = (h16*)A(FW);
  h16 *Wsa_k0 = (h16*)A(FW), *Wsa_k1 = (h16*)A(FW), *Wsa_k2 = (h16*)A(FW);
  h16 *Wsa_v = (h16*)A(FW), *Wsa_wo = (h16*)A(FW);
  h16 *Wca_k0 = (h16*)A(FW), *Wca_k1 = (h16*)A(FW), *Wca_v = (h16*)A(FW);
  h16 *Wca_q0 = (h16*)A(FW), *Wca_q1 = (h16*)A(FW), *Wca_q2 = (h16*)A(FW);
  h16 *Wca_wo = (h16*)A(FW);
  h16 *Wlin1 = (h16*)A(2048 * 256 * sizeof(h16));
  h16 *Wlin2 = (h16*)A(256 * 2048 * sizeof(h16));
  float *b_sa_q = (float*)A(1024), *b_sa_k = (float*)A(1024), *b_sa_v = (float*)A(1024);
  float *b_ca_k = (float*)A(1024), *b_ca_v = (float*)A(1024), *b_ca_q = (float*)A(1024);
  h16 *saQ = (h16*)A(1024 * 256 * 2), *saK = (h16*)A(1024 * 256 * 2), *saV = (h16*)A(1024 * 256 * 2);
  float *attn_ws = (float*)A(16 * 8 * 64 * 64 * 4);
  h16 *sa_out = (h16*)A(1024 * 256 * 2);
  float *sa_wo_out = (float*)A(1024 * 256 * 4);
  float *tgt_aF = (float*)A(1024 * 256 * 4);
  float *ca_qF = (float*)A(1024 * 256 * 4);
  h16 *ca_out = (h16*)A(512 * 256 * 2);
  float *ca_wo_out = (float*)A(512 * 256 * 4);
  float *tgt_bF = (float*)A(1024 * 256 * 4);
  h16 *tgt_bH = (h16*)A(1024 * 256 * 2);
  h16 *ffn_h = (h16*)A(1024 * 2048 * 2);
  float *ffn_out = (float*)A(1024 * 256 * 4);

  // -------- weight/bias folds
  k_fold_w<<<256, 256, 0, stream>>>(sa_wq, sa_qc_w, Wsa_q0);
  k_fold_w<<<256, 256, 0, stream>>>(sa_wq, sa_qt_w, Wsa_q1);
  k_fold_w<<<256, 256, 0, stream>>>(sa_wq, sa_qp_w, Wsa_q2);
  k_fold_w<<<256, 256, 0, stream>>>(sa_wk, sa_kc_w, Wsa_k0);
  k_fold_w<<<256, 256, 0, stream>>>(sa_wk, sa_kt_w, Wsa_k1);
  k_fold_w<<<256, 256, 0, stream>>>(sa_wk, sa_kp_w, Wsa_k2);
  k_fold_w<<<256, 256, 0, stream>>>(sa_wv, sa_v_w, Wsa_v);
  k_fold_w<<<256, 256, 0, stream>>>(ca_wk, ca_kc_w, Wca_k0);
  k_fold_w<<<256, 256, 0, stream>>>(ca_wk, ca_kp_w, Wca_k1);
  k_fold_w<<<256, 256, 0, stream>>>(ca_wv, ca_v_w, Wca_v);
  k_fold_w<<<256, 256, 0, stream>>>(ca_wq, ca_qc_w, Wca_q0);
  k_fold_w<<<256, 256, 0, stream>>>(ca_wq, ca_qs_w, Wca_q1);
  k_fold_w<<<256, 256, 0, stream>>>(ca_wq, ca_qt_w, Wca_q2);
  k_cvt_h<<<256, 256, 0, stream>>>(sa_wo, Wsa_wo, 65536);
  k_cvt_h<<<256, 256, 0, stream>>>(ca_wo, Wca_wo, 65536);
  k_cvt_h<<<2048, 256, 0, stream>>>(lin1_w, Wlin1, 2048 * 256);
  k_cvt_h<<<2048, 256, 0, stream>>>(lin2_w, Wlin2, 256 * 2048);
  k_fold_b<<<1, 256, 0, stream>>>(sa_wq, sa_qc_b, sa_bq, b_sa_q, 1);
  k_fold_b<<<1, 256, 0, stream>>>(sa_wq, sa_qt_b, nullptr, b_sa_q, 0);
  k_fold_b<<<1, 256, 0, stream>>>(sa_wq, sa_qp_b, nullptr, b_sa_q, 0);
  k_fold_b<<<1, 256, 0, stream>>>(sa_wk, sa_kc_b, sa_bk, b_sa_k, 1);
  k_fold_b<<<1, 256, 0, stream>>>(sa_wk, sa_kt_b, nullptr, b_sa_k, 0);
  k_fold_b<<<1, 256, 0, stream>>>(sa_wk, sa_kp_b, nullptr, b_sa_k, 0);
  k_fold_b<<<1, 256, 0, stream>>>(sa_wv, sa_v_b, sa_bv, b_sa_v, 1);
  k_fold_b<<<1, 256, 0, stream>>>(ca_wk, ca_kc_b, ca_bk, b_ca_k, 1);
  k_fold_b<<<1, 256, 0, stream>>>(ca_wk, ca_kp_b, nullptr, b_ca_k, 0);
  k_fold_b<<<1, 256, 0, stream>>>(ca_wv, ca_v_b, ca_bv, b_ca_v, 1);
  k_fold_b<<<1, 256, 0, stream>>>(ca_wq, ca_qc_b, ca_bq, b_ca_q, 1);
  k_fold_b<<<1, 256, 0, stream>>>(ca_wq, ca_qs_b, nullptr, b_ca_q, 0);
  k_fold_b<<<1, 256, 0, stream>>>(ca_wq, ca_qt_b, nullptr, b_ca_q, 0);

  // -------- self attention
  k_gemm<<<dim3(8, 4), 256, 0, stream>>>(tgt, qtime, qpos, Wsa_q0, Wsa_q1, Wsa_q2,
                                         3, 1024, 256, 256, b_sa_q, 0, nullptr, saQ, 0);
  k_gemm<<<dim3(8, 4), 256, 0, stream>>>(tgt, qtime, qpos, Wsa_k0, Wsa_k1, Wsa_k2,
                                         3, 1024, 256, 256, b_sa_k, 0, nullptr, saK, 0);
  k_gemm<<<dim3(8, 4), 256, 0, stream>>>(tgt, nullptr, nullptr, Wsa_v, nullptr, nullptr,
                                         1, 1024, 256, 256, b_sa_v, 0, nullptr, saV, 0);
  k_sa_attn<<<dim3(16, 8), 32, 0, stream>>>(saQ, saK, saV, attn_ws, sa_out);
  k_wred<<<256, 256, 0, stream>>>(attn_ws, (float*)d_out + 64 * 16 * 256);
  k_gemm<<<dim3(8, 4), 256, 0, stream>>>(sa_out, nullptr, nullptr, Wsa_wo, nullptr, nullptr,
                                         1, 1024, 256, 256, sa_bo, 0, sa_wo_out, nullptr, 1);
  k_add_ln<<<1024, 256, 0, stream>>>(tgt, sa_wo_out, n1_g, n1_b, tgt_aF, nullptr, nullptr, 0);

  // -------- cross attention (fused K/V projection + flash-style attention)
  k_gemm<<<dim3(8, 4), 256, 0, stream>>>(tgt_aF, qsine, qtime, Wca_q0, Wca_q1, Wca_q2,
                                         3, 1024, 256, 256, b_ca_q, 0, ca_qF, nullptr, 0);
  k_ca_attn<<<512, 256, 0, stream>>>(mem, pos, ca_qF, Wca_k0, Wca_k1, Wca_v,
                                     b_ca_k, b_ca_v, dur, ca_out);
  k_gemm<<<dim3(4, 4), 256, 0, stream>>>(ca_out, nullptr, nullptr, Wca_wo, nullptr, nullptr,
                                         1, 512, 256, 256, ca_bo, 0, ca_wo_out, nullptr, 1);
  k_add_ln<<<1024, 256, 0, stream>>>(tgt_aF, ca_wo_out, n3_g, n3_b, tgt_bF, tgt_bH, dur, 1);

  // -------- FFN
  k_gemm<<<dim3(8, 32), 256, 0, stream>>>(tgt_bH, nullptr, nullptr, Wlin1, nullptr, nullptr,
                                          1, 1024, 2048, 256, lin1_b, 1, nullptr, ffn_h, 1);
  k_gemm<<<dim3(8, 4), 256, 0, stream>>>(ffn_h, nullptr, nullptr, Wlin2, nullptr, nullptr,
                                         1, 1024, 256, 2048, lin2_b, 0, ffn_out, nullptr, 1);
  k_add_ln<<<1024, 256, 0, stream>>>(tgt_bF, ffn_out, n4_g, n4_b, (float*)d_out, nullptr,
                                     nullptr, 0);
}
